// LlamaAttention_27917287424720
// MI455X (gfx1250) — matrix operand order repaired and checked by reading, not compiled, not run on hardware
//
#include <hip/hip_runtime.h>
#include <hip/hip_bf16.h>

#define NH 32
#define NKV 8
#define HD 64

typedef __attribute__((ext_vector_type(16))) __bf16 v16bf;
typedef __attribute__((ext_vector_type(8)))  float  v8f;
typedef __attribute__((ext_vector_type(4)))  unsigned v4u;
typedef __attribute__((ext_vector_type(8)))  unsigned v8u;

union CFrag { v8f v; float f[8]; };
union AFrag { v16bf v; unsigned u[8]; };

// ---- WMMA wrapper: D = A*B + C, bf16 in, f32 acc --------------------------
__device__ __forceinline__ v8f wmma_bf16(v16bf a, v16bf b, v8f c) {
  return __builtin_amdgcn_wmma_f32_16x16x32_bf16(false, a, false, b,
                                                 (short)0, c, false, false);
}

// ---- CDNA5 async copy global->LDS (ASYNCcnt), bypasses VGPRs --------------
__device__ __forceinline__ unsigned lds_off_u32(const void* p) {
  // generic pointer to __shared__ : low 32 bits are the LDS byte offset
  return (unsigned)(unsigned long long)p;
}
__device__ __forceinline__ void async_ld_b128(unsigned lds, const void* g) {
  unsigned long long ga = (unsigned long long)g;
  asm volatile("global_load_async_to_lds_b128 %0, %1, off"
               :: "v"(lds), "v"(ga) : "memory");
}
__device__ __forceinline__ void wait_asynccnt0() {
  asm volatile("s_wait_asynccnt 0" ::: "memory");
}
__device__ __forceinline__ void wait_tensorcnt0() {
#if __has_builtin(__builtin_amdgcn_s_wait_tensorcnt)
  __builtin_amdgcn_s_wait_tensorcnt(0);
#else
  asm volatile("s_wait_tensorcnt 0" ::: "memory");
#endif
}

// ---- TDM: load 2D tile (rows x 128B) into LDS with 16B row padding --------
// tile_dim0 = 64 halfs (128B), pad_interval code 4 (=32 dwords = 128B),
// pad_amount code 3 (=4 dwords = 16B)  => LDS row stride 144B = 72 halfs.
__device__ __forceinline__ void tdm_load_tile_32x64(unsigned lds,
                                                    const void* gptr,
                                                    unsigned rows,
                                                    unsigned elem_stride) {
  unsigned long long ga = (unsigned long long)gptr;
  v4u g0;
  g0.x = 1u;                                   // count=1 (valid descriptor)
  g0.y = lds;                                  // lds_addr
  g0.z = (unsigned)(ga & 0xffffffffu);         // global_addr[31:0]
  g0.w = (unsigned)((ga >> 32) & 0x01ffffffu)  // global_addr[56:32]
         | (2u << 30);                         // type=2 ("image")
  v8u g1;
  g1.s0 = (1u << 16)     // data_size = 1 (2 bytes)
        | (1u << 20)     // pad_enable
        | (4u << 22)     // pad_interval code 4 -> every 128 bytes
        | (3u << 25);    // pad_amount code 3 -> 4 dwords (16B)
  g1.s1 = (64u & 0xffffu) << 16;               // tensor_dim0 lo16 = 64
  g1.s2 = ((rows & 0xffffu) << 16);            // dim0 hi=0 | tensor_dim1 lo16
  g1.s3 = (64u << 16);                         // dim1 hi=0 | tile_dim0 = 64
  g1.s4 = rows & 0xffffu;                      // tile_dim1 = rows, tile_dim2=0
  g1.s5 = elem_stride;                         // tensor_dim0_stride lo32
  g1.s6 = 0u;
  g1.s7 = 0u;
  asm volatile("tensor_load_to_lds %0, %1" :: "s"(g0), "s"(g1) : "memory");
}

// ---- A fragment (16x32 bf16): lanes 0-15 m=lane, K groups per VGPR --------
__device__ __forceinline__ v16bf lds_load_a(const __bf16* base, int ld) {
  const int lane = threadIdx.x & 31;
  const int m    = lane & 15;
  const int kb   = (lane & 16) >> 1;      // 0 or 8
  const __bf16* row = base + m * ld;
  AFrag f;
#pragma unroll
  for (int g = 0; g < 2; ++g)
#pragma unroll
    for (int j = 0; j < 4; ++j)
      f.u[g * 4 + j] = *(const unsigned*)(row + g * 16 + kb + 2 * j);
  return f.v;
}

// ---- B fragment (32x16 bf16) from LDS stored [n][k] -----------------------
__device__ __forceinline__ v16bf lds_load_b(const __bf16* base, int ld) {
  const int lane = threadIdx.x & 31;
  const int n    = lane & 15;
  const int kb   = (lane & 16) >> 1;      // 0 or 8
  const __bf16* col = base + n * ld + kb;
  AFrag f;
#pragma unroll
  for (int j = 0; j < 8; ++j)
    f.u[j] = *(const unsigned*)(col + (j >> 2) * 16 + 2 * (j & 3));
  return f.v;
}

// ---- row reductions across the 16-lane half holding one output row -------
__device__ __forceinline__ float redmax16(float v) {
#pragma unroll
  for (int m = 8; m; m >>= 1) v = fmaxf(v, __shfl_xor(v, m, 32));
  return v;
}
__device__ __forceinline__ float redsum16(float v) {
#pragma unroll
  for (int m = 8; m; m >>= 1) v += __shfl_xor(v, m, 32);
  return v;
}

// ---- f32 -> bf16 convert ---------------------------------------------------
__global__ void cvt_f32_bf16(const float* __restrict__ in,
                             __bf16* __restrict__ out, size_t n) {
  size_t i = (size_t)blockIdx.x * blockDim.x + threadIdx.x;
  size_t stride = (size_t)gridDim.x * blockDim.x;
  for (; i < n; i += stride) out[i] = (__bf16)in[i];
}

// ---- tiled bf16 GEMM: out[M,N] = A[M,K] @ B[K,N] --------------------------
template <bool OUTF32>
__launch_bounds__(128)
__global__ void gemm_bf16(const __bf16* __restrict__ A,
                          const __bf16* __restrict__ Bw,
                          void* __restrict__ outp, int M, int N, int K) {
  __shared__ alignas(16) __bf16 As[64 * 40];   // [row][k], stride 40
  __shared__ alignas(16) __bf16 Bs[64 * 40];   // transposed [n][k]
  const int tid  = threadIdx.x;
  const int lane = tid & 31;
  const int wid  = tid >> 5;
  const int m0 = blockIdx.y * 64, n0 = blockIdx.x * 64;
  const int wm = (wid >> 1) * 32, wn = (wid & 1) * 32;

  CFrag acc[2][2];
#pragma unroll
  for (int i = 0; i < 2; ++i)
#pragma unroll
    for (int j = 0; j < 2; ++j)
#pragma unroll
      for (int r = 0; r < 8; ++r) acc[i][j].f[r] = 0.f;

  for (int k0 = 0; k0 < K; k0 += 32) {
    __syncthreads();
    // stage A tile 64x32 via CDNA5 async b128 copies (global -> LDS direct)
#pragma unroll
    for (int i = 0; i < 2; ++i) {
      int flat = i * 128 + tid;            // 256 chunks of 8 halves (16B)
      int row  = flat >> 2;
      int ch   = (flat & 3) << 3;
      async_ld_b128(lds_off_u32(As + row * 40 + ch),
                    A + (size_t)(m0 + row) * K + k0 + ch);
    }
    // stage B tile 32x64 transposed -> Bs[n][k] (coalesced along N)
#pragma unroll
    for (int i = 0; i < 16; ++i) {
      int flat = i * 128 + tid;
      int kr = flat >> 6;
      int nc = flat & 63;
      Bs[nc * 40 + kr] = Bw[(size_t)(k0 + kr) * N + n0 + nc];
    }
    if (k0 + 32 < K)
      __builtin_prefetch(A + (size_t)(m0 + (tid >> 1)) * K + k0 + 32 + ((tid & 1) << 4), 0, 1);
    wait_asynccnt0();
    __syncthreads();

    v16bf a0 = lds_load_a(As + (wm +  0) * 40, 40);
    v16bf a1 = lds_load_a(As + (wm + 16) * 40, 40);
    v16bf b0 = lds_load_b(Bs + (wn +  0) * 40, 40);
    v16bf b1 = lds_load_b(Bs + (wn + 16) * 40, 40);
    acc[0][0].v = wmma_bf16(a0, b0, acc[0][0].v);
    acc[0][1].v = wmma_bf16(a0, b1, acc[0][1].v);
    acc[1][0].v = wmma_bf16(a1, b0, acc[1][0].v);
    acc[1][1].v = wmma_bf16(a1, b1, acc[1][1].v);
  }

  const int ncol = lane & 15;
  const int rb   = (lane & 16) >> 1;
#pragma unroll
  for (int i = 0; i < 2; ++i)
#pragma unroll
    for (int j = 0; j < 2; ++j)
#pragma unroll
      for (int r = 0; r < 8; ++r) {
        size_t row = (size_t)(m0 + wm + i * 16 + rb + r);
        size_t col = (size_t)(n0 + wn + j * 16 + ncol);
        if (OUTF32)
          ((float*)outp)[row * N + col] = acc[i][j].f[r];
        else
          ((__bf16*)outp)[row * N + col] = (__bf16)acc[i][j].f[r];
      }
}

// ---- RoPE, interleaved pairs, layout [BT][H][64] --------------------------
__global__ void rope_bf16(__bf16* __restrict__ x, int BT, int H, int T) {
  size_t n = (size_t)BT * H * 32;
  size_t stride = (size_t)gridDim.x * blockDim.x;
  for (size_t i = (size_t)blockIdx.x * blockDim.x + threadIdx.x; i < n; i += stride) {
    int pd = (int)(i & 31);
    int h  = (int)((i >> 5) % H);
    size_t bt = i / ((size_t)H * 32);
    int t = (int)(bt % T);
    size_t base = bt * (size_t)H * HD + (size_t)h * HD + 2 * pd;
    float inv_freq = __powf(10000.0f, -(float)(2 * pd) / 64.0f);
    float ang = (float)t * inv_freq;
    float c = __cosf(ang), s = __sinf(ang);
    float xe = (float)x[base], xo = (float)x[base + 1];
    x[base]     = (__bf16)(xe * c - xo * s);
    x[base + 1] = (__bf16)(xo * c + xe * s);
  }
}

// ---- flash attention: grid (T/64, NH, B), 128 threads = 4 waves -----------
__launch_bounds__(128)
__global__ void attn_fwd(const __bf16* __restrict__ q, const __bf16* __restrict__ k,
                         const __bf16* __restrict__ v, __bf16* __restrict__ y,
                         int B, int T) {
  const int C  = NH * HD;   // 2048
  const int KC = NKV * HD;  // 512
  __shared__ alignas(16) __bf16 Qs[64 * 72];       // [qrow][d]
  __shared__ alignas(16) __bf16 Ks[32 * 72];       // [key][d]  (TDM-loaded)
  __shared__ alignas(16) __bf16 Vs[64 * 40];       // transposed [d][key]
  __shared__ alignas(16) __bf16 Ps[4 * 16 * 40];   // per-wave P patch

  const int tid = threadIdx.x, lane = tid & 31, wid = tid >> 5;
  const int qb = blockIdx.x, h = blockIdx.y, b = blockIdx.z;
  const int kvh = h >> 2;   // GQA: 4 query heads per kv head

  // stage Q tile (64 rows x 64 d) via async b128 copies
  const __bf16* qbase = q + ((size_t)(b * T) + qb * 64) * C + h * HD;
#pragma unroll
  for (int i = 0; i < 4; ++i) {
    int flat = i * 128 + tid;            // 512 chunks of 8 halves
    int row = flat >> 3;
    int ch  = (flat & 7) << 3;
    async_ld_b128(lds_off_u32(Qs + row * 72 + ch),
                  qbase + (size_t)row * C + ch);
  }
  wait_asynccnt0();
  __syncthreads();

  const int m0 = wid * 16;
  v16bf qf0 = lds_load_a(Qs + m0 * 72, 72);        // d 0..31
  v16bf qf1 = lds_load_a(Qs + m0 * 72 + 32, 72);   // d 32..63

  CFrag o[4];
#pragma unroll
  for (int dt = 0; dt < 4; ++dt)
#pragma unroll
    for (int r = 0; r < 8; ++r) o[dt].f[r] = 0.f;
  float mrow[8], lrow[8];
#pragma unroll
  for (int r = 0; r < 8; ++r) { mrow[r] = -1e30f; lrow[r] = 0.f; }

  const int ncol = lane & 15;
  const int rb   = (lane & 16) >> 1;
  const int qrow0 = qb * 64 + m0;
  const int nkb = 2 * (qb + 1);

  for (int kb = 0; kb < nkb; ++kb) {
    __syncthreads();
    // stage K (32 keys x 64 d) through the Tensor Data Mover, wave 0 only;
    // TDM pads each 128B row by 16B -> LDS stride 72 halves.
    const __bf16* kbase = k + ((size_t)(b * T) + kb * 32) * KC + kvh * HD;
    if (wid == 0)
      tdm_load_tile_32x64(lds_off_u32(Ks), kbase, 32u, (unsigned)KC);
    // stage V transposed [d][key] (manual: transpose not expressible by DMA)
    const __bf16* vbase = v + ((size_t)(b * T) + kb * 32) * KC + kvh * HD;
#pragma unroll
    for (int i = 0; i < 16; ++i) {
      int flat = i * 128 + tid;
      int key = flat >> 6, d = flat & 63;
      Vs[d * 40 + key] = vbase[(size_t)key * KC + d];
    }
    if (wid == 0) wait_tensorcnt0();
    __syncthreads();

    // S = Q @ K^T  (two 16-key tiles, two 32-d k-steps each)
    CFrag s[2];
#pragma unroll
    for (int nt = 0; nt < 2; ++nt) {
#pragma unroll
      for (int r = 0; r < 8; ++r) s[nt].f[r] = 0.f;
      v16bf kf0 = lds_load_b(Ks + nt * 16 * 72, 72);
      v16bf kf1 = lds_load_b(Ks + nt * 16 * 72 + 32, 72);
      s[nt].v = wmma_bf16(qf0, kf0, s[nt].v);
      s[nt].v = wmma_bf16(qf1, kf1, s[nt].v);
    }
    // scale + causal mask
#pragma unroll
    for (int nt = 0; nt < 2; ++nt)
#pragma unroll
      for (int r = 0; r < 8; ++r) {
        int row = qrow0 + rb + r;
        int col = kb * 32 + nt * 16 + ncol;
        float val = s[nt].f[r] * 0.125f;   // 1/sqrt(64)
        s[nt].f[r] = (col <= row) ? val : -1e30f;
      }
    // online softmax update + P to LDS (wave-private patch)
    __bf16* pw = Ps + wid * 16 * 40;
#pragma unroll
    for (int r = 0; r < 8; ++r) {
      float mx = redmax16(fmaxf(s[0].f[r], s[1].f[r]));
      float mnew = fmaxf(mrow[r], mx);
      float corr = __expf(mrow[r] - mnew);
      float p0 = __expf(s[0].f[r] - mnew);
      float p1 = __expf(s[1].f[r] - mnew);
      float rs = redsum16(p0 + p1);
      lrow[r] = lrow[r] * corr + rs;
      mrow[r] = mnew;
#pragma unroll
      for (int dt = 0; dt < 4; ++dt) o[dt].f[r] *= corr;
      pw[(rb + r) * 40 + ncol]      = (__bf16)p0;
      pw[(rb + r) * 40 + 16 + ncol] = (__bf16)p1;
    }
    // O += P @ V  (C-frag -> A-frag relayout via LDS; same-wave DS in-order)
    v16bf pf = lds_load_a(pw, 40);
#pragma unroll
    for (int dt = 0; dt < 4; ++dt) {
      v16bf vf = lds_load_b(Vs + dt * 16 * 40, 40);
      o[dt].v = wmma_bf16(pf, vf, o[dt].v);
    }
  }

  // epilogue: normalize and store y[BT][C]
#pragma unroll
  for (int r = 0; r < 8; ++r) {
    float inv = 1.0f / lrow[r];
    size_t row = (size_t)(b * T) + qrow0 + rb + r;
#pragma unroll
    for (int dt = 0; dt < 4; ++dt)
      y[row * (size_t)C + h * HD + dt * 16 + ncol] = (__bf16)(o[dt].f[r] * inv);
  }
}

// ---------------------------------------------------------------------------
extern "C" void kernel_launch(void* const* d_in, const int* in_sizes, int n_in,
                              void* d_out, int out_size, void* d_ws, size_t ws_size,
                              hipStream_t stream) {
  const int B = 2, T = 2048, C = 2048, KVC = NKV * HD;   // 512
  const size_t BT = (size_t)B * T;                        // 4096

  const float* x  = (const float*)d_in[0];
  const float* wq = (const float*)d_in[1];
  const float* wk = (const float*)d_in[2];
  const float* wv = (const float*)d_in[3];
  const float* wo = (const float*)d_in[4];

  char* w = (char*)d_ws;
  __bf16* xb   = (__bf16*)w; w += BT * C * 2;
  __bf16* wqb  = (__bf16*)w; w += (size_t)C * C * 2;
  __bf16* wkb  = (__bf16*)w; w += (size_t)C * KVC * 2;
  __bf16* wvb  = (__bf16*)w; w += (size_t)C * KVC * 2;
  __bf16* wob  = (__bf16*)w; w += (size_t)C * C * 2;
  __bf16* qbuf = (__bf16*)w; w += BT * C * 2;
  __bf16* kbuf = (__bf16*)w; w += BT * KVC * 2;
  __bf16* vbuf = (__bf16*)w; w += BT * KVC * 2;
  __bf16* ybuf = (__bf16*)w; w += BT * C * 2;

  dim3 cb(256);
  cvt_f32_bf16<<<1024, cb, 0, stream>>>(x,  xb,  BT * C);
  cvt_f32_bf16<<<1024, cb, 0, stream>>>(wq, wqb, (size_t)C * C);
  cvt_f32_bf16<<<512,  cb, 0, stream>>>(wk, wkb, (size_t)C * KVC);
  cvt_f32_bf16<<<512,  cb, 0, stream>>>(wv, wvb, (size_t)C * KVC);
  cvt_f32_bf16<<<1024, cb, 0, stream>>>(wo, wob, (size_t)C * C);

  dim3 blk(128);
  dim3 gq(C / 64,  (unsigned)(BT / 64));
  dim3 gk(KVC / 64, (unsigned)(BT / 64));
  gemm_bf16<false><<<gq, blk, 0, stream>>>(xb, wqb, qbuf, (int)BT, C,   C);
  gemm_bf16<false><<<gk, blk, 0, stream>>>(xb, wkb, kbuf, (int)BT, KVC, C);
  gemm_bf16<false><<<gk, blk, 0, stream>>>(xb, wvb, vbuf, (int)BT, KVC, C);

  rope_bf16<<<1024, cb, 0, stream>>>(qbuf, (int)BT, NH,  T);
  rope_bf16<<<256,  cb, 0, stream>>>(kbuf, (int)BT, NKV, T);

  dim3 ga(T / 64, NH, B);
  attn_fwd<<<ga, blk, 0, stream>>>(qbuf, kbuf, vbuf, ybuf, B, T);

  gemm_bf16<true><<<gq, blk, 0, stream>>>(ybuf, wob, d_out, (int)BT, C, C);
}
